// BERT_83743272337839
// MI455X (gfx1250) — matrix operand order repaired and checked by reading, not compiled, not run on hardware
//
#include <hip/hip_runtime.h>

// ---------------------------------------------------------------------------
// BERT-base forward on MI455X (gfx1250).
// bf16 WMMA GEMMs (V_WMMA_F32_16X16X32_BF16, wave32), f32 accumulation.
// "BT" operand convention: B always [N,K] row-major -> every fragment is a
// pair of 16B runs in LDS (2x ds_load_b128). Double-buffered LDS pipeline with
// GLOBAL_LOAD_ASYNC_TO_LDS_B128 staging (ASYNCcnt). 128x128 block tile,
// 32x64 per wave (8 WMMAs per wave per K-step -> 1.5 LDS b128 loads / WMMA).
// ---------------------------------------------------------------------------

typedef __attribute__((ext_vector_type(16))) __bf16 v16bf;
typedef __attribute__((ext_vector_type(8)))  float  v8f;
typedef __attribute__((vector_size(16)))     int    v4i_vs;   // async builtin pointee

constexpr int BB = 8, SS = 512, HH = 768, FF = 3072, NHEAD = 12, NLAYER = 12;
constexpr int DH = 64;
constexpr int MROWS = BB * SS;       // 4096 token rows
constexpr long long LL_SS = SS;

#if __has_builtin(__builtin_amdgcn_global_load_async_to_lds_b128)
#define HAVE_ASYNC_LDS 1
#else
#define HAVE_ASYNC_LDS 0
#endif

__device__ __forceinline__ unsigned short f2bf(float f) {
  unsigned int u = __builtin_bit_cast(unsigned int, f);
  unsigned int r = u + 0x7FFFu + ((u >> 16) & 1u);   // round-to-nearest-even
  return (unsigned short)(r >> 16);
}
__device__ __forceinline__ __bf16 bfbits(unsigned short s) {
  return __builtin_bit_cast(__bf16, s);
}

#if HAVE_ASYNC_LDS
__device__ __forceinline__ void async_b128(const unsigned short* g, unsigned short* l) {
  __builtin_amdgcn_global_load_async_to_lds_b128(
      (__attribute__((address_space(1))) v4i_vs*)(g),
      (__attribute__((address_space(3))) v4i_vs*)(l), 0, 0);
}
__device__ __forceinline__ void wait_async0() {
#if __has_builtin(__builtin_amdgcn_s_wait_asynccnt)
  __builtin_amdgcn_s_wait_asynccnt(0);
#else
  asm volatile("s_wait_asynccnt 0" ::: "memory");
#endif
}
#endif

// ---------------------------------------------------------------------------
// Batched WMMA GEMM:  C[z] = act( alpha * A[z] @ B[z] + bias )
//   A : M x K bf16 row-major (lda);  BT: N x K bf16 row-major (ldbt) == B^T
//   Batch base: base = (z/bdiv)*s1 + (z%bdiv)*s2 per matrix.
// Block: 256 threads (8 wave32), tile 128(M) x 128(N), K step 32.
// Wave tile: 32(M) x 64(N) = 2x4 WMMA accumulators.
// M must be a multiple of 128; N a multiple of 16 (zero-padded to 128).
// ---------------------------------------------------------------------------
template<int DO_GELU, int WF, int WB>
__global__ __launch_bounds__(256, 1)
void wmma_gemm_bt(const unsigned short* __restrict__ A, long long lda,
                  const unsigned short* __restrict__ BT, long long ldbt,
                  const float* __restrict__ bias,
                  float* __restrict__ Cf, unsigned short* __restrict__ Cb,
                  long long ldc,
                  int M, int N, int K, float alpha,
                  int bdiv,
                  long long a_s1, long long a_s2,
                  long long b_s1, long long b_s2,
                  long long c_s1, long long c_s2)
{
  const int z = blockIdx.z;
  const long long abase = (long long)(z / bdiv) * a_s1 + (long long)(z % bdiv) * a_s2;
  const long long bbase = (long long)(z / bdiv) * b_s1 + (long long)(z % bdiv) * b_s2;
  const long long cbase = (long long)(z / bdiv) * c_s1 + (long long)(z % bdiv) * c_s2;

  const int m0 = blockIdx.y * 128;
  const int n0 = blockIdx.x * 128;

  // double-buffered tiles; 40-ushort row pitch (80B, 16B-divisible)
  __shared__ unsigned short As[2][128][40];   // A tile: 128 rows x 32 k
  __shared__ unsigned short Bs[2][128][40];   // BT tile: 128 n-rows x 32 k

  const int tid  = threadIdx.x;
  const int lane = tid & 31;
  const int w    = tid >> 5;
  const int wm   = (w & 3) * 32;   // wave M offset in block tile (2 frags)
  const int wn   = (w >> 2) * 64;  // wave N offset in block tile (4 frags)

  v8f acc[2][4];
#pragma unroll
  for (int mi = 0; mi < 2; ++mi)
#pragma unroll
    for (int nt = 0; nt < 4; ++nt)
#pragma unroll
      for (int i = 0; i < 8; ++i) acc[mi][nt][i] = 0.0f;

  // cooperative staging map (identical for A and B tiles):
  // thread t -> row t>>1 (0..127), 16 contiguous k at (t&1)*16  (two b128s)
  const int sr = tid >> 1, sc = (tid & 1) * 16;
  const bool bvalid = (n0 + sr) < N;            // N multiple of 16; zero-pad tail

  const unsigned short* Ap = A  + abase + (long long)(m0 + sr) * lda  + sc;
  const unsigned short* Bp = BT + bbase + (long long)(n0 + sr) * ldbt + sc;

  unsigned short* lA[2] = { &As[0][sr][sc], &As[1][sr][sc] };
  unsigned short* lB[2] = { &Bs[0][sr][sc], &Bs[1][sr][sc] };

#if HAVE_ASYNC_LDS
  // --- async staging: zero VGPR payload, ASYNCcnt-tracked ---
  auto stage = [&](int kt, int buf) {
    const long long k0 = (long long)kt * 32;
    async_b128(Ap + k0,     lA[buf]);
    async_b128(Ap + k0 + 8, lA[buf] + 8);
    if (bvalid) {
      async_b128(Bp + k0,     lB[buf]);
      async_b128(Bp + k0 + 8, lB[buf] + 8);
    }
  };
  auto stage_fin = [&](int buf) {
    wait_async0();
    if (!bvalid) {
      uint4 zz; zz.x = zz.y = zz.z = zz.w = 0u;
      *(uint4*)(lB[buf])     = zz;
      *(uint4*)(lB[buf] + 8) = zz;
    }
  };
#else
  // --- fallback: register staging ---
  uint4 aR0, aR1, bR0, bR1;
  auto stage = [&](int kt, int buf) {
    (void)buf;
    const long long k0 = (long long)kt * 32;
    aR0 = *(const uint4*)(Ap + k0);
    aR1 = *(const uint4*)(Ap + k0 + 8);
    if (bvalid) {
      bR0 = *(const uint4*)(Bp + k0);
      bR1 = *(const uint4*)(Bp + k0 + 8);
    } else {
      bR0.x = bR0.y = bR0.z = bR0.w = 0u;
      bR1 = bR0;
    }
  };
  auto stage_fin = [&](int buf) {
    *(uint4*)(lA[buf])     = aR0;
    *(uint4*)(lA[buf] + 8) = aR1;
    *(uint4*)(lB[buf])     = bR0;
    *(uint4*)(lB[buf] + 8) = bR1;
  };
#endif

  const int ktiles = K >> 5;
  stage(0, 0);
  stage_fin(0);
  __syncthreads();

  const int arow = wm + (lane & 15);   // + mi*16
  const int kh   = (lane >> 4) * 8;    // A frag sub-run select
  const int kb   = (lane >> 4) * 8;    // B frag sub-run select (same as A)
  const int cl   = lane & 15;

  for (int kt = 0; kt < ktiles; ++kt) {
    const int  cur  = kt & 1;
    const bool more = (kt + 1) < ktiles;
    if (more) {
      stage(kt + 1, cur ^ 1);          // safe: buffer was last read before
      if (kt + 2 < ktiles) {           // the barrier that ended iter kt-1
        __builtin_prefetch(Ap + (long long)(kt + 2) * 32, 0, 1);
        __builtin_prefetch(Bp + (long long)(kt + 2) * 32, 0, 1);
      }
    }

    // Two A fragments (rows wm..wm+15, wm+16..wm+31): 2x ds_load_b128 each
    v16bf af[2];
#pragma unroll
    for (int mi = 0; mi < 2; ++mi) {
      const unsigned short* ap = &As[cur][arow + mi * 16][0];
#pragma unroll
      for (int i = 0; i < 8; ++i) {
        af[mi][i]     = bfbits(ap[kh + i]);
        af[mi][8 + i] = bfbits(ap[16 + kh + i]);
      }
    }
    // Four B fragments; each reused by two WMMAs (the two M sub-tiles)
#pragma unroll
    for (int nt = 0; nt < 4; ++nt) {
      const unsigned short* bp = &Bs[cur][wn + nt * 16 + cl][kb];
      v16bf bfrag;
#pragma unroll
      for (int i = 0; i < 16; ++i) bfrag[i] = bfbits(bp[(i < 8) ? i : i + 8]);
#pragma unroll
      for (int mi = 0; mi < 2; ++mi)
        acc[mi][nt] = __builtin_amdgcn_wmma_f32_16x16x32_bf16(
            false, af[mi], false, bfrag, (short)0, acc[mi][nt], false, false);
    }

    if (more) { stage_fin(cur ^ 1); __syncthreads(); }
  }

  // epilogue: D layout — VGPR r: lanes 0-15 -> M=r, lanes 16-31 -> M=r+8
  const int rsel = (lane >> 4) * 8;
#pragma unroll
  for (int nt = 0; nt < 4; ++nt) {
    const int col = n0 + wn + nt * 16 + cl;
    if (col >= N) continue;
    const float bv = bias ? bias[col] : 0.0f;
#pragma unroll
    for (int mi = 0; mi < 2; ++mi) {
#pragma unroll
      for (int r = 0; r < 8; ++r) {
        const int row = m0 + wm + mi * 16 + r + rsel;
        float v = acc[mi][nt][r] * alpha + bv;
        if (DO_GELU) v = 0.5f * v * (1.0f + erff(v * 0.70710678118654752f));
        const long long idx = cbase + (long long)row * ldc + col;
        if (WF) Cf[idx] = v;
        if (WB) Cb[idx] = f2bf(v);
      }
    }
  }
}

// ---------------------------------------------------------------------------
// fp32 [K,N] -> bf16 [N,K] tiled transpose-convert (weights). block 32x8.
// ---------------------------------------------------------------------------
__global__ __launch_bounds__(256)
void cvt_transpose(const float* __restrict__ in, unsigned short* __restrict__ out,
                   int K, int N)
{
  __shared__ unsigned short tile[32][33];
  const long long mat = blockIdx.z;
  const float* src = in + mat * (long long)K * N;
  unsigned short* dst = out + mat * (long long)K * N;
  const int k0 = blockIdx.x * 32, n0 = blockIdx.y * 32;
  const int tx = threadIdx.x, ty = threadIdx.y;
#pragma unroll
  for (int i = 0; i < 32; i += 8)
    tile[ty + i][tx] = f2bf(src[(long long)(k0 + ty + i) * N + (n0 + tx)]);
  __syncthreads();
#pragma unroll
  for (int i = 0; i < 32; i += 8)
    dst[(long long)(n0 + ty + i) * K + (k0 + tx)] = tile[tx][ty + i];
}

// ---------------------------------------------------------------------------
// V [b*S, h*64+dh] bf16 -> VT [(b,h), dh, s] bf16 tiled transpose. block 32x8.
// ---------------------------------------------------------------------------
__global__ __launch_bounds__(256)
void transpose_v(const unsigned short* __restrict__ v, unsigned short* __restrict__ vt)
{
  __shared__ unsigned short tile[32][33];
  const int z = blockIdx.z;
  const int b = z / NHEAD, h = z % NHEAD;
  const int s0 = blockIdx.x * 32, d0 = blockIdx.y * 32;
  const int tx = threadIdx.x, ty = threadIdx.y;
  const unsigned short* src = v + (long long)b * SS * HH + h * DH;
  unsigned short* dst = vt + (long long)z * DH * SS;
#pragma unroll
  for (int i = 0; i < 32; i += 8)
    tile[ty + i][tx] = src[(long long)(s0 + ty + i) * HH + (d0 + tx)];
  __syncthreads();
#pragma unroll
  for (int i = 0; i < 32; i += 8)
    dst[(long long)(d0 + ty + i) * SS + (s0 + tx)] = tile[tx][ty + i];
}

// ---------------------------------------------------------------------------
// Embedding gather + sum (one block per token row)
// ---------------------------------------------------------------------------
__global__ __launch_bounds__(256)
void embed_sum(const int* __restrict__ ids, const int* __restrict__ segs,
               const float* __restrict__ tok, const float* __restrict__ pos,
               const float* __restrict__ seg, float* __restrict__ out)
{
  const long long row = blockIdx.x;
  const int t = threadIdx.x;
  const long long id = ids[row];
  const long long sg = segs[row];
  const long long s  = row % SS;
#pragma unroll
  for (int i = 0; i < 3; ++i) {
    const int c = t + i * 256;
    out[row * HH + c] = tok[id * HH + c] + pos[s * HH + c] + seg[sg * HH + c];
  }
}

// ---------------------------------------------------------------------------
// Fused (optional residual) + LayerNorm; writes f32 and bf16 activations.
// ---------------------------------------------------------------------------
__global__ __launch_bounds__(256)
void ln_residual(const float* __restrict__ in, const float* __restrict__ res,
                 const float* __restrict__ g, const float* __restrict__ b,
                 float* __restrict__ outf, unsigned short* __restrict__ outb)
{
  const long long row = blockIdx.x;
  const int t = threadIdx.x;
  __shared__ float red[256];
  float v[3];
  float s = 0.0f;
#pragma unroll
  for (int i = 0; i < 3; ++i) {
    const int c = t + i * 256;
    float x = in[row * HH + c] + (res ? res[row * HH + c] : 0.0f);
    v[i] = x; s += x;
  }
  red[t] = s; __syncthreads();
  for (int st = 128; st > 0; st >>= 1) { if (t < st) red[t] += red[t + st]; __syncthreads(); }
  const float mean = red[0] * (1.0f / HH);
  __syncthreads();
  float s2 = 0.0f;
#pragma unroll
  for (int i = 0; i < 3; ++i) { const float d = v[i] - mean; s2 += d * d; }
  red[t] = s2; __syncthreads();
  for (int st = 128; st > 0; st >>= 1) { if (t < st) red[t] += red[t + st]; __syncthreads(); }
  const float rstd = rsqrtf(red[0] * (1.0f / HH) + 1e-3f);
#pragma unroll
  for (int i = 0; i < 3; ++i) {
    const int c = t + i * 256;
    const float y = (v[i] - mean) * rstd * g[c] + b[c];
    if (outf) outf[row * HH + c] = y;
    if (outb) outb[row * HH + c] = f2bf(y);
  }
}

// ---------------------------------------------------------------------------
// Masked softmax over S=512 keys; one block (128 thr) per (b,h,q) row.
// ---------------------------------------------------------------------------
__global__ __launch_bounds__(128)
void softmax_mask(const float* __restrict__ scores, const float* __restrict__ att_mask,
                  unsigned short* __restrict__ probs)
{
  const long long row = blockIdx.x;               // over B*NH*S
  const long long bidx = row / ((long long)NHEAD * SS);
  const float* src = scores + row * LL_SS;
  unsigned short* dst = probs + row * LL_SS;
  __shared__ float red[128];
  const int t = threadIdx.x;
  float vals[4];
  float mx = -3.0e38f;
#pragma unroll
  for (int i = 0; i < 4; ++i) {
    const int k = t + i * 128;
    const float neg = (1.0f - att_mask[bidx * SS + k]) * -10000.0f;
    const float vv = src[k] + neg;
    vals[i] = vv; mx = fmaxf(mx, vv);
  }
  red[t] = mx; __syncthreads();
  for (int st = 64; st > 0; st >>= 1) { if (t < st) red[t] = fmaxf(red[t], red[t + st]); __syncthreads(); }
  mx = red[0]; __syncthreads();
  float sum = 0.0f;
#pragma unroll
  for (int i = 0; i < 4; ++i) { vals[i] = expf(vals[i] - mx); sum += vals[i]; }
  red[t] = sum; __syncthreads();
  for (int st = 64; st > 0; st >>= 1) { if (t < st) red[t] += red[t + st]; __syncthreads(); }
  const float inv = 1.0f / red[0];
#pragma unroll
  for (int i = 0; i < 4; ++i) dst[t + i * 128] = f2bf(vals[i] * inv);
}

// ---------------------------------------------------------------------------
// Host orchestration
// ---------------------------------------------------------------------------
static inline size_t alignup(size_t x) { return (x + 255) & ~(size_t)255; }

extern "C" void kernel_launch(void* const* d_in, const int* in_sizes, int n_in,
                              void* d_out, int out_size, void* d_ws, size_t ws_size,
                              hipStream_t stream) {
  (void)in_sizes; (void)n_in; (void)out_size; (void)ws_size;

  const int*   input_ids = (const int*)d_in[0];
  const int*   seg_ids   = (const int*)d_in[1];
  const float* att_mask  = (const float*)d_in[2];
  const float* tok_emb   = (const float*)d_in[3];
  const float* pos_emb   = (const float*)d_in[4];
  const float* seg_emb   = (const float*)d_in[5];
  const float* emb_g     = (const float*)d_in[6];
  const float* emb_b     = (const float*)d_in[7];
  const float* Wq = (const float*)d_in[8];  const float* bq = (const float*)d_in[9];
  const float* Wk = (const float*)d_in[10]; const float* bk = (const float*)d_in[11];
  const float* Wv = (const float*)d_in[12]; const float* bv = (const float*)d_in[13];
  const float* Wo = (const float*)d_in[14]; const float* bo = (const float*)d_in[15];
  const float* ln1_g = (const float*)d_in[16]; const float* ln1_b = (const float*)d_in[17];
  const float* Wff = (const float*)d_in[18]; const float* bff = (const float*)d_in[19];
  const float* Wo2 = (const float*)d_in[20]; const float* bo2 = (const float*)d_in[21];
  const float* ln2_g = (const float*)d_in[22]; const float* ln2_b = (const float*)d_in[23];

  // ---- workspace carve-out ----
  char* ws = (char*)d_ws;
  size_t off = 0;
  auto take = [&](size_t bytes) -> void* { void* p = ws + off; off += alignup(bytes); return p; };

  const long long nHH = (long long)NLAYER * HH * HH;
  const long long nHF = (long long)NLAYER * HH * FF;
  unsigned short* wq_b  = (unsigned short*)take(nHH * 2);   // W^T bf16
  unsigned short* wk_b  = (unsigned short*)take(nHH * 2);
  unsigned short* wv_b  = (unsigned short*)take(nHH * 2);
  unsigned short* wo_b  = (unsigned short*)take(nHH * 2);
  unsigned short* wff_b = (unsigned short*)take(nHF * 2);
  unsigned short* wo2_b = (unsigned short*)take(nHF * 2);

  const long long nAct = (long long)MROWS * HH;
  float*          x_f   = (float*)take(nAct * 4);
  unsigned short* x_b   = (unsigned short*)take(nAct * 2);
  unsigned short* q_b   = (unsigned short*)take(nAct * 2);
  unsigned short* k_b   = (unsigned short*)take(nAct * 2);
  unsigned short* v_b   = (unsigned short*)take(nAct * 2);
  unsigned short* vt_b  = (unsigned short*)take(nAct * 2);  // V^T per (b,h)
  unsigned short* ctx_b = (unsigned short*)take(nAct * 2);
  unsigned short* h_b   = (unsigned short*)take((long long)MROWS * FF * 2);
  float*          tmp_f = (float*)take(nAct * 4);

  const long long nScore = (long long)BB * NHEAD * SS * SS;
  float*          sc_f = (float*)take(nScore * 4);
  unsigned short* pr_b = (unsigned short*)take(nScore * 2);

  // ---- convert + transpose weights fp32[K,N] -> bf16[N,K] (every call) ----
  const dim3 tb(32, 8);
  cvt_transpose<<<dim3(HH / 32, HH / 32, NLAYER), tb, 0, stream>>>(Wq,  wq_b,  HH, HH);
  cvt_transpose<<<dim3(HH / 32, HH / 32, NLAYER), tb, 0, stream>>>(Wk,  wk_b,  HH, HH);
  cvt_transpose<<<dim3(HH / 32, HH / 32, NLAYER), tb, 0, stream>>>(Wv,  wv_b,  HH, HH);
  cvt_transpose<<<dim3(HH / 32, HH / 32, NLAYER), tb, 0, stream>>>(Wo,  wo_b,  HH, HH);
  cvt_transpose<<<dim3(HH / 32, FF / 32, NLAYER), tb, 0, stream>>>(Wff, wff_b, HH, FF);
  cvt_transpose<<<dim3(FF / 32, HH / 32, NLAYER), tb, 0, stream>>>(Wo2, wo2_b, FF, HH);

  // ---- embeddings + LN ----
  embed_sum<<<dim3(MROWS), dim3(256), 0, stream>>>(input_ids, seg_ids, tok_emb, pos_emb, seg_emb, tmp_f);
  ln_residual<<<dim3(MROWS), dim3(256), 0, stream>>>(tmp_f, nullptr, emb_g, emb_b, x_f, x_b);

  const dim3 gProj(HH / 128, MROWS / 128, 1);           // 6 x 32
  const dim3 gFfn1(FF / 128, MROWS / 128, 1);           // 24 x 32
  const dim3 gScore(SS / 128, SS / 128, BB * NHEAD);    // 4 x 4 x 96
  const dim3 gCtx(1, SS / 128, BB * NHEAD);             // 1 x 4 x 96
  const dim3 blk(256);

  const long long sSH = (long long)SS * HH;
  const long long sSS = (long long)SS * SS;
  const long long sDS = (long long)DH * SS;

  for (int l = 0; l < NLAYER; ++l) {
    const unsigned short* Wq_l  = wq_b  + (long long)l * HH * HH;
    const unsigned short* Wk_l  = wk_b  + (long long)l * HH * HH;
    const unsigned short* Wv_l  = wv_b  + (long long)l * HH * HH;
    const unsigned short* Wo_l  = wo_b  + (long long)l * HH * HH;
    const unsigned short* Wff_l = wff_b + (long long)l * HH * FF;
    const unsigned short* Wo2_l = wo2_b + (long long)l * FF * HH;
    const float* bq_l = bq + (long long)l * HH;
    const float* bk_l = bk + (long long)l * HH;
    const float* bv_l = bv + (long long)l * HH;
    const float* bo_l = bo + (long long)l * HH;
    const float* bff_l = bff + (long long)l * FF;
    const float* bo2_l = bo2 + (long long)l * HH;
    const float* g1 = ln1_g + (long long)l * HH; const float* b1 = ln1_b + (long long)l * HH;
    const float* g2 = ln2_g + (long long)l * HH; const float* b2 = ln2_b + (long long)l * HH;

    // Q, K, V projections -> bf16
    wmma_gemm_bt<0, 0, 1><<<gProj, blk, 0, stream>>>(
        x_b, HH, Wq_l, HH, bq_l, nullptr, q_b, HH, MROWS, HH, HH, 1.0f,
        1, 0, 0, 0, 0, 0, 0);
    wmma_gemm_bt<0, 0, 1><<<gProj, blk, 0, stream>>>(
        x_b, HH, Wk_l, HH, bk_l, nullptr, k_b, HH, MROWS, HH, HH, 1.0f,
        1, 0, 0, 0, 0, 0, 0);
    wmma_gemm_bt<0, 0, 1><<<gProj, blk, 0, stream>>>(
        x_b, HH, Wv_l, HH, bv_l, nullptr, v_b, HH, MROWS, HH, HH, 1.0f,
        1, 0, 0, 0, 0, 0, 0);

    // V -> V^T per (b,h) for the context GEMM's BT operand
    transpose_v<<<dim3(SS / 32, DH / 32, BB * NHEAD), tb, 0, stream>>>(v_b, vt_b);

    // scores = (Q @ K^T)/8 per (b,h): M=N=512, K=64. BT = K matrix [s,dh].
    wmma_gemm_bt<0, 1, 0><<<gScore, blk, 0, stream>>>(
        q_b, HH, k_b, HH, nullptr, sc_f, nullptr, SS, SS, SS, DH, 0.125f,
        NHEAD, sSH, DH, sSH, DH, (long long)NHEAD * sSS, sSS);

    // masked softmax -> bf16 probs
    softmax_mask<<<dim3(BB * NHEAD * SS), dim3(128), 0, stream>>>(sc_f, att_mask, pr_b);

    // ctx = probs @ V per (b,h): M=512, N=64, K=512. BT = V^T [64,512].
    wmma_gemm_bt<0, 0, 1><<<gCtx, blk, 0, stream>>>(
        pr_b, SS, vt_b, SS, nullptr, nullptr, ctx_b, HH, SS, DH, SS, 1.0f,
        NHEAD, (long long)NHEAD * sSS, sSS, (long long)NHEAD * sDS, sDS, sSH, DH);

    // output projection -> f32, residual + LN1
    wmma_gemm_bt<0, 1, 0><<<gProj, blk, 0, stream>>>(
        ctx_b, HH, Wo_l, HH, bo_l, tmp_f, nullptr, HH, MROWS, HH, HH, 1.0f,
        1, 0, 0, 0, 0, 0, 0);
    ln_residual<<<dim3(MROWS), dim3(256), 0, stream>>>(tmp_f, x_f, g1, b1, x_f, x_b);

    // FFN1 + exact GELU -> bf16
    wmma_gemm_bt<1, 0, 1><<<gFfn1, blk, 0, stream>>>(
        x_b, HH, Wff_l, HH, bff_l, nullptr, h_b, FF, MROWS, FF, HH, 1.0f,
        1, 0, 0, 0, 0, 0, 0);

    // FFN2 -> f32, residual + LN2
    wmma_gemm_bt<0, 1, 0><<<gProj, blk, 0, stream>>>(
        h_b, FF, Wo2_l, FF, bo2_l, tmp_f, nullptr, HH, MROWS, HH, FF, 1.0f,
        1, 0, 0, 0, 0, 0, 0);
    ln_residual<<<dim3(MROWS), dim3(256), 0, stream>>>(tmp_f, x_f, g2, b2, x_f, x_b);
  }

  (void)hipMemcpyAsync(d_out, x_f, (size_t)MROWS * HH * sizeof(float),
                       hipMemcpyDeviceToDevice, stream);
}
